// xLSTM_72189810311503
// MI455X (gfx1250) — matrix operand order repaired and checked by reading, not compiled, not run on hardware
//
#include <hip/hip_runtime.h>

#define B_   2
#define S_   2048
#define D_   768
#define V_   32000
#define NH_  4
#define DHS_ 192
#define DI_  1536
#define DHM_ 384
#define FF_  1024
#define NT_  (B_*S_)

typedef __attribute__((ext_vector_type(16))) __bf16 v16bf;
typedef __attribute__((ext_vector_type(8)))  __bf16 v8bf;
typedef __attribute__((ext_vector_type(8)))  float  v8f;

__device__ __forceinline__ v8f wmma_bf16(v16bf a, v16bf b, v8f c) {
  return __builtin_amdgcn_wmma_f32_16x16x32_bf16(false, a, false, b, (short)0, c, false, false);
}
__device__ __forceinline__ float sigmoidf_(float x) { return 1.f / (1.f + __expf(-x)); }
__device__ __forceinline__ float siluf_(float x)    { return x * sigmoidf_(x); }

__device__ __forceinline__ v16bf cat8(v8bf a, v8bf b) {
  return __builtin_shufflevector(a, b, 0,1,2,3,4,5,6,7,8,9,10,11,12,13,14,15);
}

// A 16x32 (MxK) fragment from row-major LDS tile (two contiguous 8-elem runs -> 2x ds_load_b128)
__device__ __forceinline__ v16bf frag_A_16x32(const __bf16* sm, int ld) {
  int l = threadIdx.x & 31, half = l >> 4, m = l & 15;
  const __bf16* p = sm + m * ld + half * 16;
  return cat8(*(const v8bf*)p, *(const v8bf*)(p + 8));
}
// B 32x16 fragment from fragment-major memory: lane's 16 elems contiguous
__device__ __forceinline__ v16bf frag_B_swz(const __bf16* sm) {
  const __bf16* p = sm + (threadIdx.x & 31) * 16;
  return cat8(*(const v8bf*)p, *(const v8bf*)(p + 8));
}

// Convert W[K,N] (fp32, row-major) -> fragment-major bf16:
// out[((kt*(N/16)+j)*32 + l)*16 + e] = W[kt*32 + (l>>4)*16 + e][j*16 + (l&15)]
__global__ void wswz_k(const float* __restrict__ W, __bf16* __restrict__ out, int K, int N) {
  size_t idx = (size_t)blockIdx.x * blockDim.x + threadIdx.x;
  if (idx >= (size_t)K * N) return;
  int e = (int)(idx & 15);
  int l = (int)((idx >> 4) & 31);
  size_t rest = idx >> 9;
  int nt = N >> 4;
  int j = (int)(rest % nt);
  size_t kt = rest / nt;
  size_t row = kt * 32 + (l >> 4) * 16 + e;
  int col = j * 16 + (l & 15);
  out[idx] = (__bf16)W[row * N + col];
}

// ===== bf16 WMMA GEMM: C[M,N] = A[M,K] @ Wz (+resid), Wz = swizzled bf16 weights =====
// block 256 = 8 waves; tile 128x128; wave w -> rows (w&3)*32..+31, cols (w>>2)*64..+63
// A: fp32 -> register-staged -> bf16 LDS (pipelined). B: direct global_load_b128 fragments,
// all four fragments preloaded into distinct registers so the loads clause together.
__global__ __launch_bounds__(256) void gemm_wmma(
    const float* __restrict__ A, int lda,
    const __bf16* __restrict__ Wz,
    float* __restrict__ C, const float* __restrict__ resid,
    int M, int K, int N)
{
  __shared__ __attribute__((aligned(16))) __bf16 sA[128][32];
  const int tid  = threadIdx.x;
  const int wave = tid >> 5;
  const int rowT = wave & 3;
  const int colH = wave >> 2;
  const int lane = tid & 31;
  const int bm = blockIdx.y * 128;
  const int bn = blockIdx.x * 128;
  const int nt16 = N >> 4;
  const size_t jgbase = (size_t)(blockIdx.x * 8 + colH * 4);

  v8f acc[2][4];
#pragma unroll
  for (int u = 0; u < 2; ++u)
#pragma unroll
    for (int j = 0; j < 4; ++j)
#pragma unroll
      for (int e = 0; e < 8; ++e) acc[u][j][e] = 0.f;

  // A cooperative load: thread -> row ar, 16 consecutive cols
  const int ar = tid >> 1, ac = (tid & 1) * 16;
  const float* aSrc = A + (size_t)(bm + ar) * lda + ac;

  float rAv[16];
#pragma unroll
  for (int i = 0; i < 16; ++i) rAv[i] = aSrc[i];

  for (int k0 = 0; k0 < K; k0 += 32) {
    {
      v8bf a0, a1;
#pragma unroll
      for (int i = 0; i < 8; ++i) { a0[i] = (__bf16)rAv[i]; a1[i] = (__bf16)rAv[8 + i]; }
      *(v8bf*)&sA[ar][ac] = a0;
      *(v8bf*)(&sA[ar][ac] + 8) = a1;
    }
    __syncthreads();
    // preload all four B fragments (distinct regs -> one load clause, latency hidden by WMMAs)
    const __bf16* bb = Wz + (((size_t)(k0 >> 5) * nt16) + jgbase) * 512 + lane * 16;
    v16bf fbs[4];
#pragma unroll
    for (int j = 0; j < 4; ++j) {
      const __bf16* p = bb + j * 512;
      fbs[j] = cat8(*(const v8bf*)p, *(const v8bf*)(p + 8));
    }
    // prefetch next A tile from global while WMMAs run
    int kn = k0 + 32;
    if (kn < K) {
      const float* s = aSrc + kn;
#pragma unroll
      for (int i = 0; i < 16; ++i) rAv[i] = s[i];
    }
    v16bf fa0 = frag_A_16x32(&sA[rowT * 32][0], 32);
    v16bf fa1 = frag_A_16x32(&sA[rowT * 32 + 16][0], 32);
#pragma unroll
    for (int j = 0; j < 4; ++j) {
      acc[0][j] = wmma_bf16(fa0, fbs[j], acc[0][j]);
      acc[1][j] = wmma_bf16(fa1, fbs[j], acc[1][j]);
    }
    __syncthreads();
  }

  // epilogue: uniform branch hoisted, base pointers stepped by N
  const int nn = lane & 15, rbb = (lane >> 4) * 8;
  if (resid) {
#pragma unroll
    for (int u = 0; u < 2; ++u)
#pragma unroll
      for (int j = 0; j < 4; ++j) {
        int col = bn + colH * 64 + j * 16 + nn;
        size_t base = (size_t)(bm + rowT * 32 + u * 16 + rbb) * N + col;
        float* cp = C + base;
        const float* rp = resid + base;
#pragma unroll
        for (int r = 0; r < 8; ++r) { *cp = acc[u][j][r] + *rp; cp += N; rp += N; }
      }
  } else {
#pragma unroll
    for (int u = 0; u < 2; ++u)
#pragma unroll
      for (int j = 0; j < 4; ++j) {
        int col = bn + colH * 64 + j * 16 + nn;
        float* cp = C + (size_t)(bm + rowT * 32 + u * 16 + rbb) * N + col;
#pragma unroll
        for (int r = 0; r < 8; ++r) { *cp = acc[u][j][r]; cp += N; }
      }
  }
}

// ================= small elementwise / reduction kernels =================
__global__ void embed_k(const int* __restrict__ x, const float* __restrict__ emb,
                        float* __restrict__ h) {
  size_t idx = (size_t)blockIdx.x * blockDim.x + threadIdx.x;
  if (idx >= (size_t)NT_ * D_) return;
  size_t tok = idx / D_; int d = (int)(idx % D_);
  h[idx] = emb[(size_t)x[tok] * D_ + d];
}

__global__ __launch_bounds__(256) void layernorm_rows(
    const float* __restrict__ x, const float* __restrict__ g,
    float* __restrict__ y, int D)
{
  __shared__ float red[256];
  const int tid = threadIdx.x;
  const float* row = x + (size_t)blockIdx.x * D;
  float s = 0.f;
  for (int d = tid; d < D; d += 256) s += row[d];
  red[tid] = s; __syncthreads();
  for (int st = 128; st > 0; st >>= 1) { if (tid < st) red[tid] += red[tid + st]; __syncthreads(); }
  float mu = red[0] / D; __syncthreads();
  float s2 = 0.f;
  for (int d = tid; d < D; d += 256) { float dl = row[d] - mu; s2 += dl * dl; }
  red[tid] = s2; __syncthreads();
  for (int st = 128; st > 0; st >>= 1) { if (tid < st) red[tid] += red[tid + st]; __syncthreads(); }
  float rstd = rsqrtf(red[0] / D + 1e-5f);
  float* out = y + (size_t)blockIdx.x * D;
  for (int d = tid; d < D; d += 256) out[d] = (row[d] - mu) * rstd * g[d];
}

// depthwise causal conv (K=4) + SiLU; per-batch causality
__global__ void conv_silu_k(const float* __restrict__ x, int ldx,
                            const float* __restrict__ ker,
                            float* __restrict__ y, int ldy, int C)
{
  size_t idx = (size_t)blockIdx.x * blockDim.x + threadIdx.x;
  if (idx >= (size_t)NT_ * C) return;
  int c = (int)(idx % C);
  size_t tok = idx / C;
  int s = (int)(tok % S_);
  float acc = 0.f;
#pragma unroll
  for (int j = 0; j < 4; ++j) {
    int sp = s - 3 + j;
    if (sp >= 0) acc += x[(tok + (size_t)(sp - s)) * ldx + c] * ker[c * 4 + j];
  }
  y[tok * ldy + c] = siluf_(acc);
}

// gates[tok,0..7] = concat(q,k,v)[tok,:] @ Wgate + bgate
__global__ __launch_bounds__(256) void gate_gemm(
    const float* __restrict__ q, const float* __restrict__ k, const float* __restrict__ v,
    const float* __restrict__ Wg, const float* __restrict__ bg, float* __restrict__ gates)
{
  __shared__ float red[8 * 256];
  const int tid = threadIdx.x;
  const size_t tok = blockIdx.x;
  float acc[8];
#pragma unroll
  for (int g = 0; g < 8; ++g) acc[g] = 0.f;
  for (int kk = tid; kk < 3 * DI_; kk += 256) {
    float val = (kk < DI_) ? q[tok * DI_ + kk]
              : (kk < 2 * DI_) ? k[tok * DI_ + kk - DI_]
              : v[tok * DI_ + kk - 2 * DI_];
    const float* w = Wg + (size_t)kk * 8;
#pragma unroll
    for (int g = 0; g < 8; ++g) acc[g] += val * w[g];
  }
#pragma unroll
  for (int g = 0; g < 8; ++g) red[g * 256 + tid] = acc[g];
  __syncthreads();
  for (int st = 128; st > 0; st >>= 1) {
    if (tid < st)
#pragma unroll
      for (int g = 0; g < 8; ++g) red[g * 256 + tid] += red[g * 256 + tid + st];
    __syncthreads();
  }
  if (tid < 8) gates[tok * 8 + tid] = red[tid * 256] + bg[tid];
}

// per-(b,h): a[s]=i-lfc[s+1], M[t]=cummax(a), ne[t]=exp(-lfc[t+1]-M[t])
__global__ void mlstm_precompute(const float* __restrict__ gates,
                                 float* __restrict__ aArr, float* __restrict__ Marr,
                                 float* __restrict__ neArr)
{
  int bh = threadIdx.x;
  if (bh >= B_ * NH_) return;
  int b = bh / NH_, h = bh % NH_;
  float lfc = 0.f, rm = -__builtin_inff();
  for (int s = 0; s < S_; ++s) {
    size_t tok = (size_t)b * S_ + s;
    float ip = gates[tok * 8 + h];
    float fp = gates[tok * 8 + NH_ + h];
    float lf = fminf(fp, 0.f) - log1pf(__expf(-fabsf(fp)));   // log_sigmoid
    lfc += lf;
    float a = ip - lfc;
    rm = fmaxf(rm, a);
    size_t o = (size_t)bh * S_ + s;
    aArr[o] = a; Marr[o] = rm; neArr[o] = __expf(-lfc - rm);
  }
}

// ================= mLSTM flash-style attention (WMMA) =================
// grid (B*NH, S/32), block 256 = 8 waves; wave: rs=w&1 (16 t-rows), qq=w>>1 (96 d-cols)
__global__ __launch_bounds__(256) void mlstm_attn(
    const float* __restrict__ Q, const float* __restrict__ Kt,
    const float* __restrict__ Vt, const float* __restrict__ aArr,
    const float* __restrict__ Marr, const float* __restrict__ neArr,
    float* __restrict__ O)
{
  __shared__ __attribute__((aligned(16))) __bf16 sK[32][DHM_ + 8];     // row-major (s x d)
  __shared__ __attribute__((aligned(16))) __bf16 sVs[24 * 32 * 16];    // swizzled ((j*32+l)*16+e)
  __shared__ __attribute__((aligned(16))) __bf16 sS[8][16][32];        // per-wave score staging
  __shared__ float  sa[32];

  const int bh = blockIdx.x;
  const int b = bh >> 2, h = bh & 3;
  const int t0 = blockIdx.y * 32;
  const int tid  = threadIdx.x;
  const int wave = tid >> 5;
  const int rs = wave & 1;
  const int qq = wave >> 1;
  const int l = tid & 31, half = l >> 4, ml = l & 15;
  const float scale = 0.05103103630798288f;   // 384^-0.5

  float m_r[8], ne_r[8];
#pragma unroll
  for (int r = 0; r < 8; ++r) {
    int tg = t0 + rs * 16 + half * 8 + r;
    m_r[r]  = Marr[(size_t)bh * S_ + tg];
    ne_r[r] = neArr[(size_t)bh * S_ + tg];
  }

  // Q fragments for this wave's 16 rows, K-dim = 384 (12 x 32 chunks), kept in VGPRs
  v16bf Qf[12];
  {
    const float* qrow = Q + ((size_t)(b * S_ + t0 + rs * 16 + ml)) * DI_ + h * DHM_;
#pragma unroll
    for (int i = 0; i < 12; ++i) {
      v16bf f;
#pragma unroll
      for (int ii = 0; ii < 8; ++ii) {
        int kk0 = 2 * ii;
        int kg = i * 32 + kk0 + half * 16;
        f[2*ii]   = (__bf16)qrow[kg];
        f[2*ii+1] = (__bf16)qrow[kg + 1];
      }
      Qf[i] = f;
    }
  }

  v8f acc[6];
#pragma unroll
  for (int j = 0; j < 6; ++j)
#pragma unroll
    for (int e = 0; e < 8; ++e) acc[j][e] = 0.f;
  float rsum[8];
#pragma unroll
  for (int r = 0; r < 8; ++r) rsum[r] = 0.f;

  for (int s0 = 0; s0 <= t0 + 31; s0 += 32) {
    __syncthreads();
    // K tile: row-major, 16-element contiguous runs per thread
    {
      int idx16 = tid;
#pragma unroll
      for (int p = 0; p < 3; ++p, idx16 += 256) {
        int sr = idx16 / 24, db = (idx16 % 24) * 16;
        const float* src = Kt + ((size_t)(b * S_ + s0 + sr)) * DI_ + h * DHM_ + db;
        v8bf t0, t1;
#pragma unroll
        for (int i = 0; i < 8; ++i) { t0[i] = (__bf16)src[i]; t1[i] = (__bf16)src[8 + i]; }
        *(v8bf*)&sK[sr][db] = t0;
        *(v8bf*)(&sK[sr][db] + 8) = t1;
      }
    }
    // V tile: fragment-major swizzle (16 K-consecutive values per thread, coalesced across threads)
    {
      const int rstart = (tid >> 7) * 16;
#pragma unroll
      for (int p = 0; p < 3; ++p) {
        int d = (tid & 127) + p * 128;
        const float* src = Vt + ((size_t)(b * S_ + s0 + rstart)) * DI_ + h * DHM_ + d;
        v8bf t0, t1;
#pragma unroll
        for (int i = 0; i < 8; ++i) {
          t0[i] = (__bf16)src[(size_t)i * DI_];
          t1[i] = (__bf16)src[(size_t)(i + 8) * DI_];
        }
        __bf16* dst = &sVs[(((d >> 4) * 32) + rstart + (d & 15)) * 16];
        *(v8bf*)dst = t0;
        *(v8bf*)(dst + 8) = t1;
      }
    }
    if (tid < 32) sa[tid] = aArr[(size_t)bh * S_ + s0 + tid];
    __syncthreads();

    // scores: two 16x16 tiles covering s0..s0+31; B[k][n] = K[s][k] (contiguous along k)
    v8f sc0, sc1;
#pragma unroll
    for (int e = 0; e < 8; ++e) { sc0[e] = 0.f; sc1[e] = 0.f; }
#pragma unroll
    for (int i = 0; i < 12; ++i) {
      const __bf16* pk0 = &sK[ml][i * 32 + half * 16];
      const __bf16* pk1 = &sK[16 + ml][i * 32 + half * 16];
      v16bf b0 = cat8(*(const v8bf*)pk0, *(const v8bf*)(pk0 + 8));
      v16bf b1 = cat8(*(const v8bf*)pk1, *(const v8bf*)(pk1 + 8));
      sc0 = wmma_bf16(Qf[i], b0, sc0);
      sc1 = wmma_bf16(Qf[i], b1, sc1);
    }

    // decay + causal mask + scale; stage as 16x32 A tile; accumulate row sums
    float a0 = sa[ml], a1 = sa[16 + ml];
    int sg0 = s0 + ml, sg1 = s0 + 16 + ml;
#pragma unroll
    for (int r = 0; r < 8; ++r) {
      int tg = t0 + rs * 16 + half * 8 + r;
      float v0 = (sg0 <= tg) ? sc0[r] * scale * __expf(a0 - m_r[r]) : 0.f;
      float v1 = (sg1 <= tg) ? sc1[r] * scale * __expf(a1 - m_r[r]) : 0.f;
      rsum[r] += v0 + v1;
      sS[wave][half * 8 + r][ml]      = (__bf16)v0;
      sS[wave][half * 8 + r][16 + ml] = (__bf16)v1;
    }

    v16bf af = frag_A_16x32(&sS[wave][0][0], 32);
#pragma unroll
    for (int j = 0; j < 6; ++j) {
      v16bf bf = frag_B_swz(&sVs[(qq * 6 + j) * 512]);
      acc[j] = wmma_bf16(af, bf, acc[j]);
    }
  }

  // row-sum reduction across the 16-lane half (xor masks stay in-half)
#pragma unroll
  for (int r = 0; r < 8; ++r) {
    float v = rsum[r];
    for (int msk = 1; msk < 16; msk <<= 1) v += __shfl_xor(v, msk, 32);
    rsum[r] = v;
  }
  float inv[8];
#pragma unroll
  for (int r = 0; r < 8; ++r)
    inv[r] = 1.f / (fmaxf(fabsf(rsum[r]), ne_r[r]) + 1e-6f);

#pragma unroll
  for (int j = 0; j < 6; ++j)
#pragma unroll
    for (int r = 0; r < 8; ++r) {
      int tg = t0 + rs * 16 + half * 8 + r;
      size_t idx = ((size_t)(b * S_ + tg)) * DI_ + h * DHM_ + qq * 96 + j * 16 + ml;
      O[idx] = acc[j][r] * inv[r];
    }
}

// per-head LN (DHM) + (hn + skip*xc)*silu(z)  -> in-place into att buffer
__global__ __launch_bounds__(128) void mlstm_mix(
    const float* __restrict__ att, const float* __restrict__ up,
    const float* __restrict__ xc, const float* __restrict__ gn,
    const float* __restrict__ skip, float* __restrict__ out)
{
  __shared__ float red[128];
  const int tid = threadIdx.x;
  const size_t tok = blockIdx.x;
  const int h = blockIdx.y;
  const float* a = att + tok * DI_ + h * DHM_;
  float s = 0.f;
  for (int d = tid; d < DHM_; d += 128) s += a[d];
  red[tid] = s; __syncthreads();
  for (int st = 64; st > 0; st >>= 1) { if (tid < st) red[tid] += red[tid + st]; __syncthreads(); }
  float mu = red[0] / DHM_; __syncthreads();
  float s2 = 0.f;
  for (int d = tid; d < DHM_; d += 128) { float dl = a[d] - mu; s2 += dl * dl; }
  red[tid] = s2; __syncthreads();
  for (int st = 64; st > 0; st >>= 1) { if (tid < st) red[tid] += red[tid + st]; __syncthreads(); }
  float rstd = rsqrtf(red[0] / DHM_ + 1e-5f);
  for (int d = tid; d < DHM_; d += 128) {
    int col = h * DHM_ + d;
    float hn = (a[d] - mu) * rstd * gn[col];
    float z  = up[tok * (2 * DI_) + DI_ + col];
    out[tok * DI_ + col] = (hn + skip[col] * xc[tok * DI_ + col]) * siluf_(z);
  }
}

// ================= sLSTM sequential scan: one block per (b,h), 192 threads =================
__global__ __launch_bounds__(192) void slstm_scan(
    const float* __restrict__ pre_if, const float* __restrict__ pre_zo,
    const float* __restrict__ R, const float* __restrict__ bias,
    float* __restrict__ hs)
{
  __shared__ float sh[DHS_];
  const int bh = blockIdx.x;
  const int b = bh / NH_, h = bh % NH_;
  const int e = threadIdx.x;
  sh[e] = 0.f;
  __syncthreads();
  const float bi = bias[0 * D_ + h * DHS_ + e];
  const float bf = bias[1 * D_ + h * DHS_ + e];
  const float bz = bias[2 * D_ + h * DHS_ + e];
  const float bo = bias[3 * D_ + h * DHS_ + e];
  const float* Ri = R + ((size_t)(0 * NH_ + h)) * DHS_ * DHS_ + e;
  const float* Rf = R + ((size_t)(1 * NH_ + h)) * DHS_ * DHS_ + e;
  const float* Rz = R + ((size_t)(2 * NH_ + h)) * DHS_ * DHS_ + e;
  const float* Ro = R + ((size_t)(3 * NH_ + h)) * DHS_ * DHS_ + e;
  float c = 0.f, n = 0.f, m = 0.f;
  for (int t = 0; t < S_; ++t) {
    float r0 = 0.f, r1 = 0.f, r2 = 0.f, r3 = 0.f;
    for (int d = 0; d < DHS_; ++d) {
      float hv = sh[d];
      r0 += hv * Ri[(size_t)d * DHS_];
      r1 += hv * Rf[(size_t)d * DHS_];
      r2 += hv * Rz[(size_t)d * DHS_];
      r3 += hv * Ro[(size_t)d * DHS_];
    }
    size_t tok = (size_t)b * S_ + t;
    float ib = pre_if[tok * (2 * D_) + h * DHS_ + e] + bi + r0;
    float fb = pre_if[tok * (2 * D_) + D_ + h * DHS_ + e] + bf + r1;
    float zt = tanhf(pre_zo[tok * (2 * D_) + h * DHS_ + e] + bz + r2);
    float ot = sigmoidf_(pre_zo[tok * (2 * D_) + D_ + h * DHS_ + e] + bo + r3);
    float mn = fmaxf(fb + m, ib);
    float it = __expf(ib - mn);
    float ft = __expf(fb + m - mn);
    c = ft * c + it * zt;
    n = ft * n + it;
    float hn = ot * c / fmaxf(fabsf(n), 1.f);
    m = mn;
    __syncthreads();
    sh[e] = hn;
    hs[tok * D_ + h * DHS_ + e] = hn;
    __syncthreads();
  }
}

// per-head LN (DHS) with gain, added to residual in place
__global__ __launch_bounds__(64) void slstm_ln_add(
    const float* __restrict__ hsv, const float* __restrict__ gn,
    float* __restrict__ hres)
{
  __shared__ float red[64];
  const int tid = threadIdx.x;
  const size_t tok = blockIdx.x;
  const int h = blockIdx.y;
  const float* a = hsv + tok * D_ + h * DHS_;
  float s = 0.f;
  for (int d = tid; d < DHS_; d += 64) s += a[d];
  red[tid] = s; __syncthreads();
  for (int st = 32; st > 0; st >>= 1) { if (tid < st) red[tid] += red[tid + st]; __syncthreads(); }
  float mu = red[0] / DHS_; __syncthreads();
  float s2 = 0.f;
  for (int d = tid; d < DHS_; d += 64) { float dl = a[d] - mu; s2 += dl * dl; }
  red[tid] = s2; __syncthreads();
  for (int st = 32; st > 0; st >>= 1) { if (tid < st) red[tid] += red[tid + st]; __syncthreads(); }
  float rstd = rsqrtf(red[0] / DHS_ + 1e-5f);
  for (int d = tid; d < DHS_; d += 64) {
    int col = h * DHS_ + d;
    hres[tok * D_ + col] += (a[d] - mu) * rstd * gn[col];
  }
}

// gelu(a)*g  from u[tok, 2*FF]
__global__ void gelu_mix(const float* __restrict__ u, float* __restrict__ out) {
  size_t idx = (size_t)blockIdx.x * blockDim.x + threadIdx.x;
  if (idx >= (size_t)NT_ * FF_) return;
  size_t tok = idx / FF_; int f = (int)(idx % FF_);
  float a = u[tok * (2 * FF_) + f];
  float g = u[tok * (2 * FF_) + FF_ + f];
  float t = 0.5f * a * (1.f + tanhf(0.7978845608028654f * (a + 0.044715f * a * a * a)));
  out[tok * FF_ + f] = t * g;
}

// ================= host launcher =================
extern "C" void kernel_launch(void* const* d_in, const int* in_sizes, int n_in,
                              void* d_out, int out_size, void* d_ws, size_t ws_size,
                              hipStream_t stream) {
  (void)in_sizes; (void)n_in; (void)out_size; (void)ws_size;
  const int*   x       = (const int*)  d_in[0];
  const float* emb     = (const float*)d_in[1];
  const float* ln_m    = (const float*)d_in[2];
  const float* m_Wup   = (const float*)d_in[3];
  const float* m_conv  = (const float*)d_in[4];
  const float* m_Wq    = (const float*)d_in[5];
  const float* m_Wk    = (const float*)d_in[6];
  const float* m_Wv    = (const float*)d_in[7];
  const float* m_Wgate = (const float*)d_in[8];
  const float* m_bgate = (const float*)d_in[9];
  const float* m_skip  = (const float*)d_in[10];
  const float* m_gn    = (const float*)d_in[11];
  const float* m_Wdown = (const float*)d_in[12];
  const float* ln_s    = (const float*)d_in[13];
  const float* s_conv  = (const float*)d_in[14];
  const float* s_Wif   = (const float*)d_in[15];
  const float* s_Wzo   = (const float*)d_in[16];
  const float* s_R     = (const float*)d_in[17];
  const float* s_b     = (const float*)d_in[18];
  const float* s_gn    = (const float*)d_in[19];
  const float* ln_ff   = (const float*)d_in[20];
  const float* f_W1    = (const float*)d_in[21];
  const float* f_W2    = (const float*)d_in[22];
  const float* ln_f    = (const float*)d_in[23];
  const float* head_W  = (const float*)d_in[24];
  float* out = (float*)d_out;

  float* ws = (float*)d_ws;
  const size_t NT = NT_;
  float* b_h   = ws;                      // [NT, D]   residual stream
  float* b_xn  = b_h   + NT * D_;         // [NT, D]   LN output (reused)
  float* b_up  = b_xn  + NT * D_;         // [NT, 3072] up-proj / ffn u
  float* b_xc  = b_up  + NT * 2 * DI_;    // [NT, DI]  conv out / gelu-mix
  float* b_q   = b_xc  + NT * DI_;        // [NT, DI]  q / pre_if
  float* b_k   = b_q   + NT * DI_;        // [NT, DI]  k / pre_zo
  float* b_v   = b_k   + NT * DI_;        // [NT, DI]  v / slstm hs
  float* b_att = b_v   + NT * DI_;        // [NT, DI]  attention out / mixed
  float* b_g   = b_att + NT * DI_;        // [NT, 8]
  float* b_a   = b_g   + NT * 8;          // [B*NH, S]
  float* b_M   = b_a   + (size_t)B_ * NH_ * S_;
  float* b_ne  = b_M   + (size_t)B_ * NH_ * S_;

  // bf16 swizzled weight arena (16B-aligned: all preceding sizes are multiples of 4 floats)
  __bf16* z_Wup  = (__bf16*)(b_ne + (size_t)B_ * NH_ * S_);
  __bf16* z_Wq   = z_Wup  + (size_t)D_ * 2 * DI_;
  __bf16* z_Wk   = z_Wq   + (size_t)DI_ * DI_;
  __bf16* z_Wv   = z_Wk   + (size_t)DI_ * DI_;
  __bf16* z_Wdn  = z_Wv   + (size_t)DI_ * DI_;
  __bf16* z_Wif  = z_Wdn  + (size_t)DI_ * D_;
  __bf16* z_Wzo  = z_Wif  + (size_t)D_ * 2 * D_;
  __bf16* z_W1   = z_Wzo  + (size_t)D_ * 2 * D_;
  __bf16* z_W2   = z_W1   + (size_t)D_ * 2 * FF_;
  __bf16* z_Whd  = z_W2   + (size_t)FF_ * D_;

  // 0. pre-convert all weights to fragment-major bf16 (one pass, L2/HBM-cheap)
  auto swz = [&](const float* W, __bf16* dst, int K, int N) {
    size_t total = (size_t)K * N;
    wswz_k<<<(unsigned)((total + 255) / 256), 256, 0, stream>>>(W, dst, K, N);
  };
  swz(m_Wup,  z_Wup, D_,  2 * DI_);
  swz(m_Wq,   z_Wq,  DI_, DI_);
  swz(m_Wk,   z_Wk,  DI_, DI_);
  swz(m_Wv,   z_Wv,  DI_, DI_);
  swz(m_Wdown,z_Wdn, DI_, D_);
  swz(s_Wif,  z_Wif, D_,  2 * D_);
  swz(s_Wzo,  z_Wzo, D_,  2 * D_);
  swz(f_W1,   z_W1,  D_,  2 * FF_);
  swz(f_W2,   z_W2,  FF_, D_);
  swz(head_W, z_Whd, D_,  V_);

  // 1. embedding
  embed_k<<<(NT * D_ + 255) / 256, 256, 0, stream>>>(x, emb, b_h);
  // 2. LN_m
  layernorm_rows<<<NT, 256, 0, stream>>>(b_h, ln_m, b_xn, D_);
  // 3. up projection [NT,768]@[768,3072]
  gemm_wmma<<<dim3(2 * DI_ / 128, NT / 128), 256, 0, stream>>>(b_xn, D_, z_Wup, b_up, nullptr, NT, D_, 2 * DI_);
  // 4. causal conv + silu on xm (first DI cols of up)
  conv_silu_k<<<(NT * DI_ + 255) / 256, 256, 0, stream>>>(b_up, 2 * DI_, m_conv, b_xc, DI_, DI_);
  // 5. q,k from xc ; v from xm
  gemm_wmma<<<dim3(DI_ / 128, NT / 128), 256, 0, stream>>>(b_xc, DI_,     z_Wq, b_q, nullptr, NT, DI_, DI_);
  gemm_wmma<<<dim3(DI_ / 128, NT / 128), 256, 0, stream>>>(b_xc, DI_,     z_Wk, b_k, nullptr, NT, DI_, DI_);
  gemm_wmma<<<dim3(DI_ / 128, NT / 128), 256, 0, stream>>>(b_up, 2 * DI_, z_Wv, b_v, nullptr, NT, DI_, DI_);
  // 6. gate projection (N=8)
  gate_gemm<<<NT, 256, 0, stream>>>(b_q, b_k, b_v, m_Wgate, m_bgate, b_g);
  // 7. per-head decay scans
  mlstm_precompute<<<1, 32, 0, stream>>>(b_g, b_a, b_M, b_ne);
  // 8. matrix-memory attention
  mlstm_attn<<<dim3(B_ * NH_, S_ / 32), 256, 0, stream>>>(b_q, b_k, b_v, b_a, b_M, b_ne, b_att);
  // 9. per-head LN + skip + silu(z) gating (in-place)
  mlstm_mix<<<dim3(NT, NH_), 128, 0, stream>>>(b_att, b_up, b_xc, m_gn, m_skip, b_att);
  // 10. down projection + residual
  gemm_wmma<<<dim3(D_ / 128, NT / 128), 256, 0, stream>>>(b_att, DI_, z_Wdn, b_h, b_h, NT, DI_, D_);
  // 11. sLSTM: LN, conv, projections
  layernorm_rows<<<NT, 256, 0, stream>>>(b_h, ln_s, b_xn, D_);
  conv_silu_k<<<(NT * D_ + 255) / 256, 256, 0, stream>>>(b_xn, D_, s_conv, b_xc, D_, D_);
  gemm_wmma<<<dim3(2 * D_ / 128, NT / 128), 256, 0, stream>>>(b_xc, D_, z_Wif, b_q, nullptr, NT, D_, 2 * D_);
  gemm_wmma<<<dim3(2 * D_ / 128, NT / 128), 256, 0, stream>>>(b_xn, D_, z_Wzo, b_k, nullptr, NT, D_, 2 * D_);
  // 12. recurrent scan
  slstm_scan<<<B_ * NH_, DHS_, 0, stream>>>(b_q, b_k, s_R, s_b, b_v);
  // 13. per-head LN + residual add
  slstm_ln_add<<<dim3(NT, NH_), 64, 0, stream>>>(b_v, s_gn, b_h);
  // 14. FFN
  layernorm_rows<<<NT, 256, 0, stream>>>(b_h, ln_ff, b_xn, D_);
  gemm_wmma<<<dim3(2 * FF_ / 128, NT / 128), 256, 0, stream>>>(b_xn, D_, z_W1, b_up, nullptr, NT, D_, 2 * FF_);
  gelu_mix<<<(NT * FF_ + 255) / 256, 256, 0, stream>>>(b_up, b_xc);
  gemm_wmma<<<dim3(D_ / 128, NT / 128), 256, 0, stream>>>(b_xc, FF_, z_W2, b_h, b_h, NT, FF_, D_);
  // 15. final LN + LM head
  layernorm_rows<<<NT, 256, 0, stream>>>(b_h, ln_f, b_xn, D_);
  gemm_wmma<<<dim3(V_ / 128, NT / 128), 256, 0, stream>>>(b_xn, D_, z_Whd, out, nullptr, NT, D_, V_);
}
